// AutoRegressive_84069689852496
// MI455X (gfx1250) — compile-verified, matrix operand order checked by reading, not run on hardware
//
#include <hip/hip_runtime.h>
#include <cstddef>
#include <cstdint>

// ---------------- problem constants (must match reference) ----------------
constexpr int D    = 1024;
constexpr int H    = 16;
constexpr int HD   = 64;      // D / H
constexpr int F    = 4096;
constexpr int NL   = 8;
constexpr int AVOC = 1026;
constexpr int TT   = 256;
constexpr int TA   = 1024;
constexpr int BB   = 8;
constexpr int L    = TT + TA;       // 1280
constexpr int MTOK = BB * L;        // 10240 tokens (multiple of 128)
constexpr float NEGC = -1.0e9f;

// ---------------- WMMA types ----------------
typedef __attribute__((ext_vector_type(8)))  __bf16 v8bf;
typedef __attribute__((ext_vector_type(16))) __bf16 v16bf;
typedef __attribute__((ext_vector_type(8)))  float  v8f;

__device__ __forceinline__ v8f wmma_bf16(v16bf a, v16bf b, v8f c) {
  // D(16x16,f32) = A(16x32,bf16) x B(32x16,bf16) + C
  return __builtin_amdgcn_wmma_f32_16x16x32_bf16(
      /*neg_a=*/false, a, /*neg_b=*/false, b,
      /*c_mod=*/(short)0, c, /*reuse_a=*/false, /*reuse_b=*/false);
}

// Load a 16x32 bf16 A-fragment (or equivalently a Kx16 B-fragment whose
// "rows" are the 16 output columns) from row-major memory.
// Per ISA 7.12.2: lane r=lane%16 holds row r; g=lane/16 selects K-halves;
// elems[0..7] = K (koff+g*8 .. +7), elems[8..15] = K (koff+16+g*8 .. +7).
__device__ __forceinline__ v16bf frag_ld(const __bf16* rowbase, int ld, int koff) {
  int lane = threadIdx.x & 31;
  int r = lane & 15, g = lane >> 4;
  const __bf16* p = rowbase + (size_t)r * ld + koff + g * 8;
  v8bf lo = *reinterpret_cast<const v8bf*>(p);
  v8bf hi = *reinterpret_cast<const v8bf*>(p + 16);
  v16bf o;
#pragma unroll
  for (int i = 0; i < 8; ++i) { o[i] = lo[i]; o[i + 8] = hi[i]; }
  return o;
}

// Same but with row clamping (for N not multiple of 16, e.g. vocab 1026).
__device__ __forceinline__ v16bf frag_ld_clamp(const __bf16* mat, int ld, int koff,
                                               int row0, int nrows) {
  int lane = threadIdx.x & 31;
  int r = row0 + (lane & 15);
  if (r >= nrows) r = nrows - 1;
  int g = lane >> 4;
  const __bf16* p = mat + (size_t)r * ld + koff + g * 8;
  v8bf lo = *reinterpret_cast<const v8bf*>(p);
  v8bf hi = *reinterpret_cast<const v8bf*>(p + 16);
  v16bf o;
#pragma unroll
  for (int i = 0; i < 8; ++i) { o[i] = lo[i]; o[i + 8] = hi[i]; }
  return o;
}

// ---------------- generic bf16 GEMM: C[M,N] = A[M,K] * W[N,K]^T + bias ----------------
// Block: 256 threads = 8 waves arranged 4(M) x 2(N); wave tile 32x32 (2x2 WMMA).
// Block tile 128x64.  M must be a multiple of 128 (true here: 10240).
__global__ void __launch_bounds__(256) gemm_bf16_nt(
    const __bf16* __restrict__ A, const __bf16* __restrict__ W,
    const float* __restrict__ bias, float* __restrict__ Cf, __bf16* __restrict__ Cb,
    int N, int K, int relu)
{
  int wid  = threadIdx.x >> 5;
  int lane = threadIdx.x & 31;
  int wm = wid & 3, wn = wid >> 2;
  int m0 = blockIdx.y * 128 + wm * 32;
  int n0 = blockIdx.x * 64  + wn * 32;
  const __bf16* Am = A + (size_t)m0 * K;

  v8f acc00 = {}, acc01 = {}, acc10 = {}, acc11 = {};
  for (int k = 0; k < K; k += 32) {
    v16bf a0 = frag_ld(Am, K, k);
    v16bf a1 = frag_ld(Am + (size_t)16 * K, K, k);
    v16bf b0 = frag_ld_clamp(W, K, k, n0, N);
    v16bf b1 = frag_ld_clamp(W, K, k, n0 + 16, N);
    acc00 = wmma_bf16(a0, b0, acc00);
    acc01 = wmma_bf16(a0, b1, acc01);
    acc10 = wmma_bf16(a1, b0, acc10);
    acc11 = wmma_bf16(a1, b1, acc11);
  }

  int g = lane >> 4, nl = lane & 15;
  v8f accs[2][2] = { {acc00, acc01}, {acc10, acc11} };
#pragma unroll
  for (int mi = 0; mi < 2; ++mi) {
#pragma unroll
    for (int ni = 0; ni < 2; ++ni) {
      int n = n0 + ni * 16 + nl;
      if (n >= N) continue;
      float bv = bias ? bias[n] : 0.0f;
#pragma unroll
      for (int i = 0; i < 8; ++i) {
        int m = m0 + mi * 16 + i + 8 * g;
        float v = accs[mi][ni][i] + bv;
        if (relu && v < 0.0f) v = 0.0f;
        size_t idx = (size_t)m * N + n;
        if (Cf) Cf[idx] = v;
        if (Cb) Cb[idx] = (__bf16)v;
      }
    }
  }
}

// ---------------- flash attention ----------------
// qkv layout: (token, 3*D) bf16, q at col h*64, k at D+h*64, v at 2D+h*64.
// Vt layout: (b, h, d, t) bf16  (row = head dim, ld = L) for contiguous PV B-frags.
// Block = 128 threads = 4 waves; each wave owns 16 query rows.
__global__ void __launch_bounds__(128) attn_kernel(
    const __bf16* __restrict__ qkv, const __bf16* __restrict__ Vt,
    const int* __restrict__ tlen, const int* __restrict__ alen,
    __bf16* __restrict__ O)
{
  __shared__ __align__(16) __bf16 pshm[4][16 * 32];
  int wid = threadIdx.x >> 5, lane = threadIdx.x & 31;
  int g = lane >> 4, nl = lane & 15;

  const int nq = L / 64;                      // 20 q-blocks of 64 rows
  int qblk = blockIdx.x % nq;
  int h    = (blockIdx.x / nq) % H;
  int b    = blockIdx.x / (nq * H);
  int q0   = qblk * 64 + wid * 16;
  int tot  = tlen[b] + alen[b];

  const __bf16* qbase = qkv + ((size_t)(b * L + q0)) * (3 * D) + h * HD;
  const __bf16* kbase = qkv + ((size_t)(b * L)) * (3 * D) + D + h * HD;
  const __bf16* vbase = Vt + ((size_t)(b * H + h)) * HD * L;

  v16bf qa0 = frag_ld(qbase, 3 * D, 0);    // head dims 0..31
  v16bf qa1 = frag_ld(qbase, 3 * D, 32);   // head dims 32..63

  float mrow[8], lrow[8];
#pragma unroll
  for (int i = 0; i < 8; ++i) { mrow[i] = -3.0e38f; lrow[i] = 0.0f; }
  v8f o0 = {}, o1 = {}, o2 = {}, o3 = {};
  const float scale = 0.125f;              // 1/sqrt(64)

  int kc_end = ((q0 + 16 + 31) / 32) * 32; // causal upper bound (tot handled by mask)
  for (int kc = 0; kc < kc_end; kc += 32) {
    // ---- S = Q K^T for keys [kc, kc+32) ----
    v8f s0 = {}, s1 = {};
    {
      const __bf16* kb0 = kbase + (size_t)kc * (3 * D);
      const __bf16* kb1 = kbase + (size_t)(kc + 16) * (3 * D);
      s0 = wmma_bf16(qa0, frag_ld(kb0, 3 * D, 0),  s0);
      s0 = wmma_bf16(qa1, frag_ld(kb0, 3 * D, 32), s0);
      s1 = wmma_bf16(qa0, frag_ld(kb1, 3 * D, 0),  s1);
      s1 = wmma_bf16(qa1, frag_ld(kb1, 3 * D, 32), s1);
    }
    // ---- mask + online softmax (rows m = i + 8*g, cols = kc + nt*16 + nl) ----
    float p0[8], p1[8], tmax[8];
#pragma unroll
    for (int i = 0; i < 8; ++i) {
      int qpos = q0 + i + 8 * g;
      int k0p = kc + nl, k1p = kc + 16 + nl;
      float v0 = s0[i] * scale; if (k0p > qpos || k0p >= tot) v0 += NEGC;
      float v1 = s1[i] * scale; if (k1p > qpos || k1p >= tot) v1 += NEGC;
      p0[i] = v0; p1[i] = v1;
      tmax[i] = fmaxf(v0, v1);
    }
#pragma unroll
    for (int m = 1; m < 16; m <<= 1) {
#pragma unroll
      for (int i = 0; i < 8; ++i) tmax[i] = fmaxf(tmax[i], __shfl_xor(tmax[i], m, 32));
    }
    float rs[8];
#pragma unroll
    for (int i = 0; i < 8; ++i) {
      float mn = fmaxf(mrow[i], tmax[i]);
      float alpha = __expf(mrow[i] - mn);
      mrow[i] = mn;
      float e0 = __expf(p0[i] - mn);
      float e1 = __expf(p1[i] - mn);
      rs[i] = e0 + e1;
      lrow[i] *= alpha;
      o0[i] *= alpha; o1[i] *= alpha; o2[i] *= alpha; o3[i] *= alpha;
      // C-layout -> row-major LDS (16x32 bf16) for A-fragment reload
      pshm[wid][(i + 8 * g) * 32 + nl]      = (__bf16)e0;
      pshm[wid][(i + 8 * g) * 32 + 16 + nl] = (__bf16)e1;
    }
#pragma unroll
    for (int m = 1; m < 16; m <<= 1) {
#pragma unroll
      for (int i = 0; i < 8; ++i) rs[i] += __shfl_xor(rs[i], m, 32);
    }
#pragma unroll
    for (int i = 0; i < 8; ++i) lrow[i] += rs[i];

    // LDS ops are in-order within a wave; fence the compiler + wait counter.
    asm volatile("s_wait_dscnt 0" ::: "memory");
    v16bf pa = frag_ld(&pshm[wid][0], 32, 0);

    // ---- O += P V ; B-frag rows are head dims (contiguous in Vt, ld=L) ----
    o0 = wmma_bf16(pa, frag_ld(vbase + (size_t)0  * L, L, kc), o0);
    o1 = wmma_bf16(pa, frag_ld(vbase + (size_t)16 * L, L, kc), o1);
    o2 = wmma_bf16(pa, frag_ld(vbase + (size_t)32 * L, L, kc), o2);
    o3 = wmma_bf16(pa, frag_ld(vbase + (size_t)48 * L, L, kc), o3);
  }

  // ---- epilogue: normalize and store bf16 (token, D) ----
  __bf16* orow = O + ((size_t)(b * L + q0)) * D + h * HD;
#pragma unroll
  for (int i = 0; i < 8; ++i) {
    float inv = 1.0f / lrow[i];
    size_t r = (size_t)(i + 8 * g) * D;
    orow[r + 0  + nl] = (__bf16)(o0[i] * inv);
    orow[r + 16 + nl] = (__bf16)(o1[i] * inv);
    orow[r + 32 + nl] = (__bf16)(o2[i] * inv);
    orow[r + 48 + nl] = (__bf16)(o3[i] * inv);
  }
}

// ---------------- V transpose: qkv(v part) -> Vt (b,h,d,t) ----------------
__global__ void vtrans_kernel(const __bf16* __restrict__ qkv, __bf16* __restrict__ Vt) {
  size_t idx = (size_t)blockIdx.x * 256 + threadIdx.x;
  if (idx >= (size_t)MTOK * D) return;
  int c = (int)(idx % D);
  size_t bt = idx / D;
  int t = (int)(bt % L);
  int b = (int)(bt / L);
  int h = c >> 6, d = c & 63;
  Vt[(((size_t)(b * H + h)) * HD + d) * L + t] = qkv[bt * (3 * D) + 2 * D + c];
}

// ---------------- residual + LayerNorm (in-place on x), emits f32 + bf16 ----------------
__global__ void __launch_bounds__(256) resid_ln_kernel(
    const float* __restrict__ y, const float* __restrict__ sc, const float* __restrict__ bi,
    float* __restrict__ x, __bf16* __restrict__ xb)
{
  __shared__ float red[17];
  size_t base = (size_t)blockIdx.x * D;
  float v[4]; float sum = 0.0f, sq = 0.0f;
#pragma unroll
  for (int j = 0; j < 4; ++j) {
    int c = threadIdx.x + j * 256;
    float a = x[base + c] + y[base + c];
    v[j] = a; sum += a; sq += a * a;
  }
#pragma unroll
  for (int m = 1; m < 32; m <<= 1) { sum += __shfl_xor(sum, m, 32); sq += __shfl_xor(sq, m, 32); }
  int wid = threadIdx.x >> 5, lane = threadIdx.x & 31;
  if (lane == 0) { red[wid] = sum; red[8 + wid] = sq; }
  __syncthreads();
  if (wid == 0) {
    float s2 = (lane < 8) ? red[lane] : 0.0f;
    float q2 = (lane < 8) ? red[8 + lane] : 0.0f;
#pragma unroll
    for (int m = 1; m < 8; m <<= 1) { s2 += __shfl_xor(s2, m, 32); q2 += __shfl_xor(q2, m, 32); }
    if (lane == 0) { red[0] = s2; red[8] = q2; }
  }
  __syncthreads();
  float mean = red[0] * (1.0f / D);
  float var  = red[8] * (1.0f / D) - mean * mean;
  float rstd = rsqrtf(var + 1e-5f);
#pragma unroll
  for (int j = 0; j < 4; ++j) {
    int c = threadIdx.x + j * 256;
    float o = (v[j] - mean) * rstd * sc[c] + bi[c];
    x[base + c] = o;
    xb[base + c] = (__bf16)o;
  }
}

// ---------------- embedding + sinusoidal PE + packing/masking ----------------
__global__ void __launch_bounds__(256) embed_kernel(
    const int* __restrict__ text, const int* __restrict__ audio,
    const int* __restrict__ tlen, const int* __restrict__ alen,
    const float* __restrict__ temb, const float* __restrict__ aemb,
    float* __restrict__ x, __bf16* __restrict__ xb)
{
  int tok = blockIdx.x;
  int b = tok / L, t = tok % L;
  int tl = tlen[b];
  int tot = tl + alen[b];
#pragma unroll
  for (int j = 0; j < 4; ++j) {
    int c = threadIdx.x + j * 256;
    float val = 0.0f;
    if (t < tot) {
      int pos; const float* emb;
      if (t < tl) {
        pos = t;
        emb = temb + (size_t)text[b * TT + t] * D;
      } else {
        int ai = t - tl; if (ai > TA - 1) ai = TA - 1;
        pos = ai;
        emb = aemb + (size_t)audio[b * TA + ai] * D;
      }
      int jj = c >> 1;
      float div = __expf(-9.2103403719761836f * (float)(2 * jj) / (float)D);
      float ang = (float)pos * div;
      float pe = (c & 1) ? cosf(ang) : sinf(ang);
      val = emb[c] + pe;
    }
    x[(size_t)tok * D + c] = val;
    xb[(size_t)tok * D + c] = (__bf16)val;
  }
}

// ---------------- f32 -> bf16 conversion ----------------
__global__ void cvt_bf16_kernel(const float* __restrict__ in, __bf16* __restrict__ out, size_t n) {
  size_t i = (size_t)blockIdx.x * 256 + threadIdx.x;
  size_t stride = (size_t)gridDim.x * 256;
  for (; i < n; i += stride) out[i] = (__bf16)in[i];
}

// ---------------- host orchestration ----------------
extern "C" void kernel_launch(void* const* d_in, const int* in_sizes, int n_in,
                              void* d_out, int out_size, void* d_ws, size_t ws_size,
                              hipStream_t stream) {
  (void)in_sizes; (void)n_in; (void)out_size; (void)ws_size;
  const int*   text  = (const int*)d_in[0];
  const int*   audio = (const int*)d_in[1];
  const int*   tlen  = (const int*)d_in[2];
  const int*   alen  = (const int*)d_in[3];
  const float* temb  = (const float*)d_in[4];
  const float* aemb  = (const float*)d_in[5];
  const float* Wqkv  = (const float*)d_in[6];
  const float* bqkv  = (const float*)d_in[7];
  const float* Wo    = (const float*)d_in[8];
  const float* bo    = (const float*)d_in[9];
  const float* W1    = (const float*)d_in[10];
  const float* b1    = (const float*)d_in[11];
  const float* W2    = (const float*)d_in[12];
  const float* b2    = (const float*)d_in[13];
  const float* ln1s  = (const float*)d_in[14];
  const float* ln1b  = (const float*)d_in[15];
  const float* ln2s  = (const float*)d_in[16];
  const float* ln2b  = (const float*)d_in[17];

  char* ws = (char*)d_ws;
  size_t off = 0;
  auto take = [&](size_t bytes) -> void* {
    void* p = ws + off;
    off = (off + bytes + 255) & ~(size_t)255;
    return p;
  };
  __bf16* xb    = (__bf16*)take((size_t)MTOK * D * 2);
  float*  xf    = (float*) take((size_t)MTOK * D * 4);
  __bf16* qkvb  = (__bf16*)take((size_t)MTOK * 3 * D * 2);
  __bf16* vtb   = (__bf16*)take((size_t)MTOK * D * 2);
  __bf16* ob    = (__bf16*)take((size_t)MTOK * D * 2);
  float*  yf    = (float*) take((size_t)MTOK * D * 4);
  __bf16* hb    = (__bf16*)take((size_t)MTOK * F * 2);
  __bf16* wqkvb = (__bf16*)take((size_t)NL * 3 * D * D * 2);
  __bf16* wob   = (__bf16*)take((size_t)NL * D * D * 2);
  __bf16* w1b   = (__bf16*)take((size_t)NL * F * D * 2);
  __bf16* w2b   = (__bf16*)take((size_t)NL * D * F * 2);
  __bf16* aembb = (__bf16*)take((size_t)AVOC * D * 2);

  // weight conversions (deterministic every call)
  auto cvt = [&](const float* src, __bf16* dst, size_t n) {
    int blocks = (int)((n + 255) / 256);
    if (blocks > 32768) blocks = 32768;
    cvt_bf16_kernel<<<blocks, 256, 0, stream>>>(src, dst, n);
  };
  cvt(Wqkv, wqkvb, (size_t)NL * 3 * D * D);
  cvt(Wo,   wob,   (size_t)NL * D * D);
  cvt(W1,   w1b,   (size_t)NL * F * D);
  cvt(W2,   w2b,   (size_t)NL * D * F);
  cvt(aemb, aembb, (size_t)AVOC * D);

  embed_kernel<<<MTOK, 256, 0, stream>>>(text, audio, tlen, alen, temb, aemb, xf, xb);

  const dim3 blk(256);
  for (int l = 0; l < NL; ++l) {
    // qkv = x @ Wqkv[l].T + bqkv[l]          (10240 x 3072)
    gemm_bf16_nt<<<dim3(3 * D / 64, MTOK / 128), blk, 0, stream>>>(
        xb, wqkvb + (size_t)l * 3 * D * D, bqkv + (size_t)l * 3 * D,
        nullptr, qkvb, 3 * D, D, 0);
    // V transpose for PV B-fragments
    vtrans_kernel<<<(int)(((size_t)MTOK * D + 255) / 256), blk, 0, stream>>>(qkvb, vtb);
    // flash attention -> ob (bf16, token x D)
    attn_kernel<<<BB * H * (L / 64), dim3(128), 0, stream>>>(qkvb, vtb, tlen, alen, ob);
    // o @ Wo.T + bo -> yf (f32)
    gemm_bf16_nt<<<dim3(D / 64, MTOK / 128), blk, 0, stream>>>(
        ob, wob + (size_t)l * D * D, bo + (size_t)l * D,
        yf, nullptr, D, D, 0);
    // x = LN(x + yf)
    resid_ln_kernel<<<MTOK, blk, 0, stream>>>(yf, ln1s + (size_t)l * D, ln1b + (size_t)l * D, xf, xb);
    // h = relu(x @ W1.T + b1)  (bf16)
    gemm_bf16_nt<<<dim3(F / 64, MTOK / 128), blk, 0, stream>>>(
        xb, w1b + (size_t)l * F * D, b1 + (size_t)l * F,
        nullptr, hb, F, D, 1);
    // y = h @ W2.T + b2  (f32)
    gemm_bf16_nt<<<dim3(D / 64, MTOK / 128), blk, 0, stream>>>(
        hb, w2b + (size_t)l * D * F, b2 + (size_t)l * D,
        yf, nullptr, D, F, 0);
    // x = LN(x + yf)
    resid_ln_kernel<<<MTOK, blk, 0, stream>>>(yf, ln2s + (size_t)l * D, ln2b + (size_t)l * D, xf, xb);
  }

  // logits = x @ audio_emb.T  (10240 x 1026, f32 to d_out)
  gemm_bf16_nt<<<dim3((AVOC + 63) / 64, MTOK / 128), blk, 0, stream>>>(
      xb, aembb, nullptr, (float*)d_out, nullptr, AVOC, D, 0);
}
